// CustomEncoderBlock_84069689852158
// MI455X (gfx1250) — matrix operand order repaired and checked by reading, not compiled, not run on hardware
//
#include <hip/hip_runtime.h>
#include <math.h>

// ---------------------------------------------------------------------------
// CDNA5 / gfx1250 implementation of the CustomEncoderBlock reference.
// Strategy: one generic bf16-WMMA GEMM (fp32 accumulate) for every matmul,
// small fused fp32 kernels for LayerNorm / RMS / softmax / assembly.
// Workspace requirement: ~640 MB in d_ws.
// ---------------------------------------------------------------------------

typedef __bf16 bf16_t;
typedef __attribute__((ext_vector_type(16))) __bf16 v16bf;
typedef __attribute__((ext_vector_type(8)))  __bf16 v8bf;
typedef __attribute__((ext_vector_type(8)))  float  v8f;

#define D_MODEL 1024
#define N_HEAD  16
#define HDIM    64
#define B_SZ    64
#define S_POOL  512
#define T_COND  11
#define M_X     (B_SZ * T_COND)   // 704
#define M_POOL  (B_SZ * S_POOL)   // 32768

// ---------------------------------------------------------------------------
// helpers
// ---------------------------------------------------------------------------
__device__ inline v8bf zero8() {
    v8bf v;
#pragma unroll
    for (int i = 0; i < 8; ++i) v[i] = (bf16_t)0.0f;
    return v;
}

__device__ inline float gelu_exact(float x) {
    return 0.5f * x * (1.0f + erff(x * 0.70710678118654752f));
}

// ---------------------------------------------------------------------------
// fp32 -> bf16 grid-stride convert
// ---------------------------------------------------------------------------
__global__ void k_cvt_bf16(const float* __restrict__ src, bf16_t* __restrict__ dst,
                           long n) {
    long i = (long)blockIdx.x * blockDim.x + threadIdx.x;
    long stride = (long)gridDim.x * blockDim.x;
    for (; i < n; i += stride) dst[i] = (bf16_t)src[i];
}

// ---------------------------------------------------------------------------
// Generic GEMM:  C[M,N] = A[M,K](bf16) * W[N,K](bf16)^T + bias [+gelu] [+res]
// 256 threads, 8 waves, 128x128 tile, BK=32. Single-buffered LDS staging.
// LDS row stride = 40 halfs (80B) -> 16B aligned for ds_load_b128,
// conflict-free across a 16-lane group.
// ---------------------------------------------------------------------------
#define LSTR 40

__global__ __launch_bounds__(256)
void k_gemm_wmma(const bf16_t* __restrict__ A, const bf16_t* __restrict__ W,
                 const float* __restrict__ bias, const float* __restrict__ res,
                 float* __restrict__ Cf, bf16_t* __restrict__ Cb,
                 int M, int N, int K, int epi_gelu) {
    __shared__ __align__(16) bf16_t sA[128 * LSTR];
    __shared__ __align__(16) bf16_t sB[128 * LSTR];

    const int tid  = threadIdx.x;
    const int wave = tid >> 5;
    const int lane = tid & 31;
    const int m0 = blockIdx.x * 128;
    const int n0 = blockIdx.y * 128;
    const int wm = (wave & 1) * 64;    // wave row offset within tile
    const int wn = (wave >> 1) * 32;   // wave col offset within tile
    const int lr = lane & 15;
    const int lh = lane >> 4;

    v8f acc[4][2];
#pragma unroll
    for (int i = 0; i < 4; ++i)
#pragma unroll
        for (int j = 0; j < 2; ++j)
#pragma unroll
            for (int e = 0; e < 8; ++e) acc[i][j][e] = 0.0f;

    for (int k0 = 0; k0 < K; k0 += 32) {
        // --- stage 128x32 A tile and 128x32 W tile into LDS (bf16) ---
        int s = tid;
#pragma unroll
        for (int it = 0; it < 2; ++it) {
            int row = s >> 2;       // 0..127
            int seg = s & 3;        // 16B segment within the 64B row
            int gr = m0 + row;
            v8bf va = zero8();
            if (gr < M) va = *(const v8bf*)(A + (size_t)gr * K + k0 + seg * 8);
            *(v8bf*)(sA + row * LSTR + seg * 8) = va;
            int gn = n0 + row;
            v8bf vb = zero8();
            if (gn < N) vb = *(const v8bf*)(W + (size_t)gn * K + k0 + seg * 8);
            *(v8bf*)(sB + row * LSTR + seg * 8) = vb;
            s += 256;
        }
        __syncthreads();

        // --- A fragments: lane<16 holds K0-7 (+0B) & K16-23 (+32B);
        //     lane>=16 holds K8-15 & K24-31 (ISA 16-bit A layout) ---
        v16bf afrag[4];
#pragma unroll
        for (int i = 0; i < 4; ++i) {
            const bf16_t* p = sA + (wm + i * 16 + lr) * LSTR + lh * 8;
            v8bf lo = *(const v8bf*)p;
            v8bf hi = *(const v8bf*)(p + 16);
            afrag[i] = __builtin_shufflevector(lo, hi, 0, 1, 2, 3, 4, 5, 6, 7,
                                               8, 9, 10, 11, 12, 13, 14, 15);
        }
        // --- B fragments: same K layout as A (K 8h..8h+7 & 16+8h..16+8h+7) ---
        v16bf bfrag[2];
#pragma unroll
        for (int j = 0; j < 2; ++j) {
            const bf16_t* p = sB + (wn + j * 16 + lr) * LSTR + lh * 8;
            v8bf lo = *(const v8bf*)p;
            v8bf hi = *(const v8bf*)(p + 16);
            bfrag[j] = __builtin_shufflevector(lo, hi, 0, 1, 2, 3, 4, 5, 6, 7,
                                               8, 9, 10, 11, 12, 13, 14, 15);
        }

#pragma unroll
        for (int i = 0; i < 4; ++i)
#pragma unroll
            for (int j = 0; j < 2; ++j)
                acc[i][j] = __builtin_amdgcn_wmma_f32_16x16x32_bf16(
                    false, afrag[i], false, bfrag[j], (short)0, acc[i][j],
                    false, false);
        __syncthreads();
    }

    // --- epilogue: f32 C/D layout (lane<16: M=v, lane>=16: M=v+8; N=lane&15)
#pragma unroll
    for (int i = 0; i < 4; ++i)
#pragma unroll
        for (int j = 0; j < 2; ++j) {
            int gn = n0 + wn + j * 16 + lr;
#pragma unroll
            for (int e = 0; e < 8; ++e) {
                int gm = m0 + wm + i * 16 + lh * 8 + e;
                if (gm < M && gn < N) {
                    float x = acc[i][j][e];
                    if (bias) x += bias[gn];
                    if (epi_gelu) x = gelu_exact(x);
                    if (res) x += res[(size_t)gm * N + gn];
                    size_t o = (size_t)gm * N + gn;
                    if (Cf) Cf[o] = x;
                    if (Cb) Cb[o] = (bf16_t)x;
                }
            }
        }
}

// ---------------------------------------------------------------------------
// LayerNorm over D=1024 (one block per row, 256 threads x 4 cols).
// Optional fp32 and/or bf16 outputs.
// ---------------------------------------------------------------------------
__global__ __launch_bounds__(256)
void k_ln(const float* __restrict__ in, const float* __restrict__ w,
          const float* __restrict__ b, float* __restrict__ outf,
          bf16_t* __restrict__ outb, int R) {
    int row = blockIdx.x;
    if (row >= R) return;
    int tid = threadIdx.x;
    const float4 x = ((const float4*)(in + (size_t)row * D_MODEL))[tid];
    __shared__ float red[256];
    red[tid] = x.x + x.y + x.z + x.w;
    __syncthreads();
    for (int o = 128; o > 0; o >>= 1) {
        if (tid < o) red[tid] += red[tid + o];
        __syncthreads();
    }
    float mu = red[0] * (1.0f / D_MODEL);
    __syncthreads();
    float d0 = x.x - mu, d1 = x.y - mu, d2 = x.z - mu, d3 = x.w - mu;
    red[tid] = d0 * d0 + d1 * d1 + d2 * d2 + d3 * d3;
    __syncthreads();
    for (int o = 128; o > 0; o >>= 1) {
        if (tid < o) red[tid] += red[tid + o];
        __syncthreads();
    }
    float rs = rsqrtf(red[0] * (1.0f / D_MODEL) + 1e-5f);
    int c = tid * 4;
    const float4 wv = ((const float4*)w)[tid];
    const float4 bv = ((const float4*)b)[tid];
    float y0 = d0 * rs * wv.x + bv.x;
    float y1 = d1 * rs * wv.y + bv.y;
    float y2 = d2 * rs * wv.z + bv.z;
    float y3 = d3 * rs * wv.w + bv.w;
    if (outf) {
        float4 o4 = {y0, y1, y2, y3};
        ((float4*)(outf + (size_t)row * D_MODEL))[tid] = o4;
    }
    if (outb) {
        bf16_t* op = outb + (size_t)row * D_MODEL + c;
        op[0] = (bf16_t)y0; op[1] = (bf16_t)y1;
        op[2] = (bf16_t)y2; op[3] = (bf16_t)y3;
    }
}

// ---------------------------------------------------------------------------
// LayerNorm + per-head RMS (for pool K path): ln_out = LN(x) (bf16),
// kn_out = rms_norm(heads(LN(x)), kn_w) (bf16). One block per row.
// ---------------------------------------------------------------------------
__global__ __launch_bounds__(256)
void k_ln_rms(const float* __restrict__ in, const float* __restrict__ lnw,
              const float* __restrict__ lnb, const float* __restrict__ knw,
              bf16_t* __restrict__ ln_out, bf16_t* __restrict__ kn_out, int R) {
    int row = blockIdx.x;
    if (row >= R) return;
    int tid = threadIdx.x;
    const float4 x = ((const float4*)(in + (size_t)row * D_MODEL))[tid];
    __shared__ float red[256];
    __shared__ float rsh[N_HEAD];
    red[tid] = x.x + x.y + x.z + x.w;
    __syncthreads();
    for (int o = 128; o > 0; o >>= 1) {
        if (tid < o) red[tid] += red[tid + o];
        __syncthreads();
    }
    float mu = red[0] * (1.0f / D_MODEL);
    __syncthreads();
    float d0 = x.x - mu, d1 = x.y - mu, d2 = x.z - mu, d3 = x.w - mu;
    red[tid] = d0 * d0 + d1 * d1 + d2 * d2 + d3 * d3;
    __syncthreads();
    for (int o = 128; o > 0; o >>= 1) {
        if (tid < o) red[tid] += red[tid + o];
        __syncthreads();
    }
    float rs = rsqrtf(red[0] * (1.0f / D_MODEL) + 1e-5f);
    __syncthreads();
    int c = tid * 4;
    const float4 wv = ((const float4*)lnw)[tid];
    const float4 bv = ((const float4*)lnb)[tid];
    float y0 = d0 * rs * wv.x + bv.x;
    float y1 = d1 * rs * wv.y + bv.y;
    float y2 = d2 * rs * wv.z + bv.z;
    float y3 = d3 * rs * wv.w + bv.w;
    // per-head (64-dim) RMS of the LN output
    red[tid] = y0 * y0 + y1 * y1 + y2 * y2 + y3 * y3;
    __syncthreads();
    if (tid < N_HEAD) {
        float t = 0.0f;
        for (int i = 0; i < 16; ++i) t += red[tid * 16 + i];
        rsh[tid] = rsqrtf(t * (1.0f / HDIM) + 1e-6f);
    }
    __syncthreads();
    float hr = rsh[tid >> 4];
    bf16_t* lp = ln_out + (size_t)row * D_MODEL + c;
    bf16_t* kp = kn_out + (size_t)row * D_MODEL + c;
    lp[0] = (bf16_t)y0; lp[1] = (bf16_t)y1; lp[2] = (bf16_t)y2; lp[3] = (bf16_t)y3;
    kp[0] = (bf16_t)(y0 * hr * knw[(c + 0) & 63]);
    kp[1] = (bf16_t)(y1 * hr * knw[(c + 1) & 63]);
    kp[2] = (bf16_t)(y2 * hr * knw[(c + 2) & 63]);
    kp[3] = (bf16_t)(y3 * hr * knw[(c + 3) & 63]);
}

// ---------------------------------------------------------------------------
// Per-head RMS of the (batch-shared) pool query -> qn (fp32, 1024). 1 block.
// ---------------------------------------------------------------------------
__global__ __launch_bounds__(256)
void k_rms_query(const float* __restrict__ pq, const float* __restrict__ qnw,
                 float* __restrict__ qn) {
    int tid = threadIdx.x;
    const float4 x = ((const float4*)pq)[tid];
    __shared__ float red[256];
    __shared__ float rsh[N_HEAD];
    red[tid] = x.x * x.x + x.y * x.y + x.z * x.z + x.w * x.w;
    __syncthreads();
    if (tid < N_HEAD) {
        float t = 0.0f;
        for (int i = 0; i < 16; ++i) t += red[tid * 16 + i];
        rsh[tid] = rsqrtf(t * (1.0f / HDIM) + 1e-6f);
    }
    __syncthreads();
    float hr = rsh[tid >> 4];
    int c = tid * 4;
    qn[c + 0] = x.x * hr * qnw[(c + 0) & 63];
    qn[c + 1] = x.y * hr * qnw[(c + 1) & 63];
    qn[c + 2] = x.z * hr * qnw[(c + 2) & 63];
    qn[c + 3] = x.w * hr * qnw[(c + 3) & 63];
}

// ---------------------------------------------------------------------------
// qproj[n] = qn . wq[n,:] + bq[n]  (wq = first D rows of packed in_w, fp32)
// grid(4) x 256 threads.
// ---------------------------------------------------------------------------
__global__ __launch_bounds__(256)
void k_qproj(const float* __restrict__ qn, const float* __restrict__ in_w,
             const float* __restrict__ in_b, float* __restrict__ qproj) {
    int n = blockIdx.x * 256 + threadIdx.x;
    const float* wrow = in_w + (size_t)n * D_MODEL;
    float a = 0.0f;
    for (int k = 0; k < D_MODEL; ++k) a += qn[k] * wrow[k];
    qproj[n] = a + in_b[n];
}

// ---------------------------------------------------------------------------
// Pooled attention: one block per (b,h); q is the shared projected query.
// scores over S=512 keys, softmax, weighted sum of V -> o (bf16).
// ---------------------------------------------------------------------------
__global__ __launch_bounds__(256)
void k_pool_attn(const float* __restrict__ qproj, const bf16_t* __restrict__ K,
                 const bf16_t* __restrict__ V, bf16_t* __restrict__ O) {
    int b = blockIdx.x >> 4;
    int h = blockIdx.x & 15;
    int tid = threadIdx.x;
    __shared__ float q[HDIM];
    __shared__ float sc[S_POOL];
    __shared__ float red[256];
    if (tid < HDIM) q[tid] = qproj[h * HDIM + tid];
    __syncthreads();
    for (int s = tid; s < S_POOL; s += 256) {
        const bf16_t* kp = K + ((size_t)(b * S_POOL + s)) * D_MODEL + h * HDIM;
        float d = 0.0f;
#pragma unroll
        for (int i = 0; i < HDIM; ++i) d += q[i] * (float)kp[i];
        sc[s] = d * 0.125f;  // 1/sqrt(64)
    }
    __syncthreads();
    float m = -1e30f;
    for (int s = tid; s < S_POOL; s += 256) m = fmaxf(m, sc[s]);
    red[tid] = m;
    __syncthreads();
    for (int o = 128; o > 0; o >>= 1) {
        if (tid < o) red[tid] = fmaxf(red[tid], red[tid + o]);
        __syncthreads();
    }
    float mx = red[0];
    __syncthreads();
    float ls = 0.0f;
    for (int s = tid; s < S_POOL; s += 256) {
        float e = expf(sc[s] - mx);
        sc[s] = e;
        ls += e;
    }
    red[tid] = ls;
    __syncthreads();
    for (int o = 128; o > 0; o >>= 1) {
        if (tid < o) red[tid] += red[tid + o];
        __syncthreads();
    }
    float inv = 1.0f / red[0];
    __syncthreads();
    if (tid < HDIM) {
        float acc = 0.0f;
        for (int s = 0; s < S_POOL; ++s)
            acc += sc[s] * (float)V[((size_t)(b * S_POOL + s)) * D_MODEL + h * HDIM + tid];
        O[(size_t)b * D_MODEL + h * HDIM + tid] = (bf16_t)(acc * inv);
    }
}

// ---------------------------------------------------------------------------
// 11-token self-attention: one block per (b,h), reads packed qkv (fp32),
// writes concatenated head outputs (bf16).
// ---------------------------------------------------------------------------
__global__ __launch_bounds__(128)
void k_attn11(const float* __restrict__ qkv, bf16_t* __restrict__ out) {
    int b = blockIdx.x >> 4;
    int h = blockIdx.x & 15;
    int tid = threadIdx.x;
    __shared__ float q[T_COND][HDIM], k[T_COND][HDIM], v[T_COND][HDIM];
    __shared__ float p[T_COND][T_COND + 1];
    for (int idx = tid; idx < T_COND * HDIM; idx += 128) {
        int t = idx >> 6, d = idx & 63;
        size_t base = ((size_t)(b * T_COND + t)) * (3 * D_MODEL) + h * HDIM + d;
        q[t][d] = qkv[base];
        k[t][d] = qkv[base + D_MODEL];
        v[t][d] = qkv[base + 2 * D_MODEL];
    }
    __syncthreads();
    for (int idx = tid; idx < T_COND * T_COND; idx += 128) {
        int qi = idx / T_COND, ki = idx % T_COND;
        float s = 0.0f;
#pragma unroll
        for (int d = 0; d < HDIM; ++d) s += q[qi][d] * k[ki][d];
        p[qi][ki] = s * 0.125f;
    }
    __syncthreads();
    if (tid < T_COND) {
        float mx = -1e30f;
        for (int ki = 0; ki < T_COND; ++ki) mx = fmaxf(mx, p[tid][ki]);
        float sm = 0.0f;
        for (int ki = 0; ki < T_COND; ++ki) {
            float e = expf(p[tid][ki] - mx);
            p[tid][ki] = e;
            sm += e;
        }
        float inv = 1.0f / sm;
        for (int ki = 0; ki < T_COND; ++ki) p[tid][ki] *= inv;
    }
    __syncthreads();
    for (int idx = tid; idx < T_COND * HDIM; idx += 128) {
        int qi = idx >> 6, d = idx & 63;
        float a = 0.0f;
#pragma unroll
        for (int ki = 0; ki < T_COND; ++ki) a += p[qi][ki] * v[ki][d];
        out[((size_t)(b * T_COND + qi)) * D_MODEL + h * HDIM + d] = (bf16_t)a;
    }
}

// ---------------------------------------------------------------------------
// Assemble x = LN(concat(t_emb, pv, pr, cond_e) + pos_emb). Block per row.
// ---------------------------------------------------------------------------
__global__ __launch_bounds__(256)
void k_assemble(const float* __restrict__ ts, const float* __restrict__ pv,
                const float* __restrict__ pr, const float* __restrict__ conde,
                const float* __restrict__ pos, const float* __restrict__ w,
                const float* __restrict__ b, float* __restrict__ x) {
    int row = blockIdx.x;            // b*11 + t
    int bb = row / T_COND, t = row % T_COND;
    int tid = threadIdx.x;
    int c = tid * 4;
    float val[4];
    if (t == 0) {
        float tv = ts[bb];
        float scale = logf(10000.0f) / 511.0f;
#pragma unroll
        for (int i = 0; i < 4; ++i) {
            int ci = c + i;
            if (ci < 512) val[i] = sinf(tv * expf(-scale * (float)ci));
            else          val[i] = cosf(tv * expf(-scale * (float)(ci - 512)));
        }
    } else if (t == 1) {
#pragma unroll
        for (int i = 0; i < 4; ++i) val[i] = pv[(size_t)bb * D_MODEL + c + i];
    } else if (t == 2) {
#pragma unroll
        for (int i = 0; i < 4; ++i) val[i] = pr[(size_t)bb * D_MODEL + c + i];
    } else {
#pragma unroll
        for (int i = 0; i < 4; ++i)
            val[i] = conde[((size_t)(bb * 8 + t - 3)) * D_MODEL + c + i];
    }
#pragma unroll
    for (int i = 0; i < 4; ++i) val[i] += pos[(size_t)t * D_MODEL + c + i];

    __shared__ float red[256];
    red[tid] = val[0] + val[1] + val[2] + val[3];
    __syncthreads();
    for (int o = 128; o > 0; o >>= 1) {
        if (tid < o) red[tid] += red[tid + o];
        __syncthreads();
    }
    float mu = red[0] * (1.0f / D_MODEL);
    __syncthreads();
    float d0 = val[0] - mu, d1 = val[1] - mu, d2 = val[2] - mu, d3 = val[3] - mu;
    red[tid] = d0 * d0 + d1 * d1 + d2 * d2 + d3 * d3;
    __syncthreads();
    for (int o = 128; o > 0; o >>= 1) {
        if (tid < o) red[tid] += red[tid + o];
        __syncthreads();
    }
    float rs = rsqrtf(red[0] * (1.0f / D_MODEL) + 1e-5f);
    float* xp = x + (size_t)row * D_MODEL + c;
    xp[0] = d0 * rs * w[c + 0] + b[c + 0];
    xp[1] = d1 * rs * w[c + 1] + b[c + 1];
    xp[2] = d2 * rs * w[c + 2] + b[c + 2];
    xp[3] = d3 * rs * w[c + 3] + b[c + 3];
}

// ---------------------------------------------------------------------------
// host-side helpers
// ---------------------------------------------------------------------------
static void cvt(hipStream_t s, const float* src, bf16_t* dst, long n) {
    int grid = (int)((n + 255) / 256);
    if (grid > 4096) grid = 4096;
    k_cvt_bf16<<<grid, 256, 0, s>>>(src, dst, n);
}

static void gemm(hipStream_t s, const bf16_t* A, const bf16_t* W,
                 const float* bias, const float* res, float* Cf, bf16_t* Cb,
                 int M, int N, int K, int gelu) {
    dim3 g((M + 127) / 128, N / 128);
    k_gemm_wmma<<<g, 256, 0, s>>>(A, W, bias, res, Cf, Cb, M, N, K, gelu);
}

extern "C" void kernel_launch(void* const* d_in, const int* in_sizes, int n_in,
                              void* d_out, int out_size, void* d_ws,
                              size_t ws_size, hipStream_t stream) {
    (void)in_sizes; (void)n_in; (void)out_size; (void)ws_size;
    // -------- inputs (setup_inputs dict order, params flattened in insertion order)
    const float* ts   = (const float*)d_in[0];
    const float* cond = (const float*)d_in[1];
    const float* vlf  = (const float*)d_in[2];
    const float* rf   = (const float*)d_in[3];
    int pi = 4;
#define PARAM(name) const float* name = (const float*)d_in[pi++]
    PARAM(cond_obs_w); PARAM(cond_obs_b);
    PARAM(vl_proj_w);  PARAM(vl_proj_b);
    PARAM(vl_ln_w);    PARAM(vl_ln_b);
    PARAM(r_proj_w);   PARAM(r_proj_b);
    PARAM(r_ln_w);     PARAM(r_ln_b);
    PARAM(vl_pq); PARAM(vl_qn_w); PARAM(vl_kn_w);
    PARAM(vl_in_w); PARAM(vl_in_b); PARAM(vl_out_w); PARAM(vl_out_b);
    PARAM(r_pq);  PARAM(r_qn_w);  PARAM(r_kn_w);
    PARAM(r_in_w);  PARAM(r_in_b);  PARAM(r_out_w);  PARAM(r_out_b);
    PARAM(pos_emb); PARAM(pre_ln_w); PARAM(pre_ln_b);
    const float *L_in_w[4], *L_in_b[4], *L_out_w[4], *L_out_b[4];
    const float *L_l1_w[4], *L_l1_b[4], *L_l2_w[4], *L_l2_b[4];
    const float *L_ln1_w[4], *L_ln1_b[4], *L_ln2_w[4], *L_ln2_b[4];
    for (int l = 0; l < 4; ++l) {
        L_in_w[l]  = (const float*)d_in[pi++]; L_in_b[l]  = (const float*)d_in[pi++];
        L_out_w[l] = (const float*)d_in[pi++]; L_out_b[l] = (const float*)d_in[pi++];
        L_l1_w[l]  = (const float*)d_in[pi++]; L_l1_b[l]  = (const float*)d_in[pi++];
        L_l2_w[l]  = (const float*)d_in[pi++]; L_l2_b[l]  = (const float*)d_in[pi++];
        L_ln1_w[l] = (const float*)d_in[pi++]; L_ln1_b[l] = (const float*)d_in[pi++];
        L_ln2_w[l] = (const float*)d_in[pi++]; L_ln2_b[l] = (const float*)d_in[pi++];
    }
    PARAM(mem_ln_w); PARAM(mem_ln_b);
    PARAM(mp1_w); PARAM(mp1_b); PARAM(mp2_w); PARAM(mp2_b);
#undef PARAM

    // -------- workspace bump allocator (~640 MB)
    char* base = (char*)d_ws;
    size_t off = 0;
    auto alloc = [&](size_t bytes) -> char* {
        size_t o = (off + 255) & ~(size_t)255;
        off = o + bytes;
        return base + o;
    };
    // bf16 weights
    bf16_t* wb_cond   = (bf16_t*)alloc((size_t)1024 * 512 * 2);
    bf16_t* wb_vlproj = (bf16_t*)alloc((size_t)1024 * 1536 * 2);
    bf16_t* wb_rproj  = (bf16_t*)alloc((size_t)1024 * 1536 * 2);
    bf16_t* wb_vl_in  = (bf16_t*)alloc((size_t)3072 * 1024 * 2);
    bf16_t* wb_vl_out = (bf16_t*)alloc((size_t)1024 * 1024 * 2);
    bf16_t* wb_r_in   = (bf16_t*)alloc((size_t)3072 * 1024 * 2);
    bf16_t* wb_r_out  = (bf16_t*)alloc((size_t)1024 * 1024 * 2);
    bf16_t *wb_Lin[4], *wb_Lout[4], *wb_Ll1[4], *wb_Ll2[4];
    for (int l = 0; l < 4; ++l) {
        wb_Lin[l]  = (bf16_t*)alloc((size_t)3072 * 1024 * 2);
        wb_Lout[l] = (bf16_t*)alloc((size_t)1024 * 1024 * 2);
        wb_Ll1[l]  = (bf16_t*)alloc((size_t)4096 * 1024 * 2);
        wb_Ll2[l]  = (bf16_t*)alloc((size_t)1024 * 4096 * 2);
    }
    bf16_t* wb_mp1 = (bf16_t*)alloc((size_t)1024 * 1024 * 2);
    bf16_t* wb_mp2 = (bf16_t*)alloc((size_t)1024 * 1024 * 2);
    // activations (reused between vl and r streams)
    bf16_t* feat_bf = (bf16_t*)alloc((size_t)M_POOL * 1536 * 2);
    float*  proj_f  = (float*)alloc((size_t)M_POOL * D_MODEL * 4);
    bf16_t* ln_bf   = (bf16_t*)alloc((size_t)M_POOL * D_MODEL * 2);
    bf16_t* kn_bf   = (bf16_t*)alloc((size_t)M_POOL * D_MODEL * 2);
    bf16_t* k_bf    = (bf16_t*)alloc((size_t)M_POOL * D_MODEL * 2);
    bf16_t* v_bf    = (bf16_t*)alloc((size_t)M_POOL * D_MODEL * 2);
    bf16_t* o_bf    = (bf16_t*)alloc((size_t)B_SZ * D_MODEL * 2);
    float*  pv_f    = (float*)alloc((size_t)B_SZ * D_MODEL * 4);
    float*  pr_f    = (float*)alloc((size_t)B_SZ * D_MODEL * 4);
    float*  qn_f    = (float*)alloc((size_t)D_MODEL * 4);
    float*  qproj_f = (float*)alloc((size_t)D_MODEL * 4);
    bf16_t* cond_bf = (bf16_t*)alloc((size_t)512 * 512 * 2);
    float*  conde_f = (float*)alloc((size_t)512 * D_MODEL * 4);
    float*  xA      = (float*)alloc((size_t)M_X * D_MODEL * 4);
    float*  xB      = (float*)alloc((size_t)M_X * D_MODEL * 4);
    bf16_t* h_bf    = (bf16_t*)alloc((size_t)M_X * D_MODEL * 2);
    float*  qkv_f   = (float*)alloc((size_t)M_X * 3 * D_MODEL * 4);
    bf16_t* attn_bf = (bf16_t*)alloc((size_t)M_X * D_MODEL * 2);
    bf16_t* mlp_bf  = (bf16_t*)alloc((size_t)M_X * 4096 * 2);

    // -------- weight conversion
    cvt(stream, cond_obs_w, wb_cond,   (long)1024 * 512);
    cvt(stream, vl_proj_w,  wb_vlproj, (long)1024 * 1536);
    cvt(stream, r_proj_w,   wb_rproj,  (long)1024 * 1536);
    cvt(stream, vl_in_w,    wb_vl_in,  (long)3072 * 1024);
    cvt(stream, vl_out_w,   wb_vl_out, (long)1024 * 1024);
    cvt(stream, r_in_w,     wb_r_in,   (long)3072 * 1024);
    cvt(stream, r_out_w,    wb_r_out,  (long)1024 * 1024);
    for (int l = 0; l < 4; ++l) {
        cvt(stream, L_in_w[l],  wb_Lin[l],  (long)3072 * 1024);
        cvt(stream, L_out_w[l], wb_Lout[l], (long)1024 * 1024);
        cvt(stream, L_l1_w[l],  wb_Ll1[l],  (long)4096 * 1024);
        cvt(stream, L_l2_w[l],  wb_Ll2[l],  (long)1024 * 4096);
    }
    cvt(stream, mp1_w, wb_mp1, (long)1024 * 1024);
    cvt(stream, mp2_w, wb_mp2, (long)1024 * 1024);

    // -------- cond token embedding: (512,512) @ (1024,512)^T
    cvt(stream, cond, cond_bf, (long)512 * 512);
    gemm(stream, cond_bf, wb_cond, cond_obs_b, nullptr, conde_f, nullptr,
         512, 1024, 512, 0);

    // -------- pooled streams (vl then reasoning; buffers reused)
    struct PoolSpec {
        const float *feat, *proj_w_unused, *proj_b, *ln_w, *ln_b, *pq, *qn_w, *kn_w;
        const float *in_w, *in_b, *out_b;
        const bf16_t *wb_proj, *wb_in, *wb_out;
        float* pooled;
    };
    PoolSpec specs[2] = {
        {vlf, vl_proj_w, vl_proj_b, vl_ln_w, vl_ln_b, vl_pq, vl_qn_w, vl_kn_w,
         vl_in_w, vl_in_b, vl_out_b, wb_vlproj, wb_vl_in, wb_vl_out, pv_f},
        {rf,  r_proj_w,  r_proj_b,  r_ln_w,  r_ln_b,  r_pq,  r_qn_w,  r_kn_w,
         r_in_w,  r_in_b,  r_out_b,  wb_rproj,  wb_r_in,  wb_r_out,  pr_f}};
    for (int sIdx = 0; sIdx < 2; ++sIdx) {
        PoolSpec& sp = specs[sIdx];
        cvt(stream, sp.feat, feat_bf, (long)M_POOL * 1536);
        // projection to D, fp32
        gemm(stream, feat_bf, sp.wb_proj, sp.proj_b, nullptr, proj_f, nullptr,
             M_POOL, D_MODEL, 1536, 0);
        // LN + per-head RMS -> bf16 operands
        k_ln_rms<<<M_POOL, 256, 0, stream>>>(proj_f, sp.ln_w, sp.ln_b, sp.kn_w,
                                             ln_bf, kn_bf, M_POOL);
        // shared query: rms then q-projection (fp32 GEMV, tiny)
        k_rms_query<<<1, 256, 0, stream>>>(sp.pq, sp.qn_w, qn_f);
        k_qproj<<<4, 256, 0, stream>>>(qn_f, sp.in_w, sp.in_b, qproj_f);
        // K = kn @ wk^T, V = ln @ wv^T (packed in_proj rows D..2D, 2D..3D)
        gemm(stream, kn_bf, sp.wb_in + (size_t)1024 * 1024, sp.in_b + 1024,
             nullptr, nullptr, k_bf, M_POOL, D_MODEL, D_MODEL, 0);
        gemm(stream, ln_bf, sp.wb_in + (size_t)2048 * 1024, sp.in_b + 2048,
             nullptr, nullptr, v_bf, M_POOL, D_MODEL, D_MODEL, 0);
        // single-query attention + out projection
        k_pool_attn<<<B_SZ * N_HEAD, 256, 0, stream>>>(qproj_f, k_bf, v_bf, o_bf);
        gemm(stream, o_bf, sp.wb_out, sp.out_b, nullptr, sp.pooled, nullptr,
             B_SZ, D_MODEL, D_MODEL, 0);
    }

    // -------- assemble x = pre_ln(concat + pos_emb)
    k_assemble<<<M_X, 256, 0, stream>>>(ts, pv_f, pr_f, conde_f, pos_emb,
                                        pre_ln_w, pre_ln_b, xA);

    // -------- 4 pre-norm transformer layers on (704, 1024)
    for (int l = 0; l < 4; ++l) {
        k_ln<<<M_X, 256, 0, stream>>>(xA, L_ln1_w[l], L_ln1_b[l], nullptr,
                                      h_bf, M_X);
        gemm(stream, h_bf, wb_Lin[l], L_in_b[l], nullptr, qkv_f, nullptr,
             M_X, 3 * D_MODEL, D_MODEL, 0);
        k_attn11<<<B_SZ * N_HEAD, 128, 0, stream>>>(qkv_f, attn_bf);
        gemm(stream, attn_bf, wb_Lout[l], L_out_b[l], xA, xB, nullptr,
             M_X, D_MODEL, D_MODEL, 0);
        k_ln<<<M_X, 256, 0, stream>>>(xB, L_ln2_w[l], L_ln2_b[l], nullptr,
                                      h_bf, M_X);
        gemm(stream, h_bf, wb_Ll1[l], L_l1_b[l], nullptr, nullptr, mlp_bf,
             M_X, 4096, D_MODEL, 1 /*gelu*/);
        gemm(stream, mlp_bf, wb_Ll2[l], L_l2_b[l], xB, xA, nullptr,
             M_X, D_MODEL, 4096, 0);
    }

    // -------- memory head: LN -> gelu(mp1) -> mp2 -> d_out (fp32)
    k_ln<<<M_X, 256, 0, stream>>>(xA, mem_ln_w, mem_ln_b, nullptr, h_bf, M_X);
    gemm(stream, h_bf, wb_mp1, mp1_b, nullptr, nullptr, mlp_bf,
         M_X, D_MODEL, D_MODEL, 1 /*gelu*/);
    gemm(stream, mlp_bf, wb_mp2, mp2_b, nullptr, (float*)d_out, nullptr,
         M_X, D_MODEL, D_MODEL, 0);
}
